// GemmaAttention_5308579578448
// MI455X (gfx1250) — matrix operand order repaired and checked by reading, not compiled, not run on hardware
//
#include <hip/hip_runtime.h>

// ---------------------------------------------------------------------------
// GemmaAttention for MI455X (gfx1250)
//   - all matmuls via v_wmma_f32_16x16x32_bf16 (fp32 accumulate)
//   - Q tile staged to LDS via Tensor Data Mover (tensor_load_to_lds)
//   - 200KB LDS working set per attention block (320KB/WGP available)
// ---------------------------------------------------------------------------

#define DEV __device__ __forceinline__

typedef __attribute__((ext_vector_type(16))) __bf16 bf16x16;
typedef __attribute__((ext_vector_type(8)))  float  f32x8;
typedef __attribute__((ext_vector_type(4)))  unsigned int u32x4;
typedef __attribute__((ext_vector_type(8)))  int    i32x8;
typedef __attribute__((ext_vector_type(4)))  int    i32x4;

union FragB { bf16x16 v; uint4 q[2]; };

static constexpr int B_    = 2;
static constexpr int S_    = 2048;
static constexpr int HID_  = 2048;
static constexpr int NH_   = 16;
static constexpr int NKV_  = 4;
static constexpr int HD_   = 256;
static constexpr int M_    = B_ * S_;        // 4096
static constexpr int NQ_   = NH_ * HD_;      // 4096
static constexpr int NKVD_ = NKV_ * HD_;     // 1024

#if defined(__has_builtin)
#if __has_builtin(__builtin_amdgcn_tensor_load_to_lds) && __has_builtin(__builtin_amdgcn_s_wait_tensorcnt)
#define HAVE_TDM 1
#endif
#endif
#ifndef HAVE_TDM
#define HAVE_TDM 0
#endif

DEV unsigned short f2bf(float f) {
  unsigned u = __float_as_uint(f);
  u += 0x7FFFu + ((u >> 16) & 1u);           // round-to-nearest-even
  return (unsigned short)(u >> 16);
}
DEV float bf2f(unsigned short h) { return __uint_as_float(((unsigned)h) << 16); }

// ---------------------------------------------------------------------------
// Prep kernels
// ---------------------------------------------------------------------------
__global__ __launch_bounds__(256) void k_f32_to_bf16(const float* __restrict__ in,
                                                     unsigned short* __restrict__ out,
                                                     long long n) {
  long long stride = (long long)gridDim.x * blockDim.x;
  for (long long i = (long long)blockIdx.x * blockDim.x + threadIdx.x; i < n; i += stride)
    out[i] = f2bf(in[i]);
}

// in: fp32 [K, N] row-major  ->  out: bf16 [N, K] row-major
__global__ __launch_bounds__(256) void k_transpose_f32_bf16(const float* __restrict__ in,
                                                            unsigned short* __restrict__ out,
                                                            int K, int N) {
  long long n = (long long)K * N;
  long long stride = (long long)gridDim.x * blockDim.x;
  for (long long i = (long long)blockIdx.x * blockDim.x + threadIdx.x; i < n; i += stride) {
    int k = (int)(i / N);
    int c = (int)(i % N);
    out[(size_t)c * K + k] = f2bf(in[i]);
  }
}

// RoPE in place on bf16 [b, s, heads, 256]; one thread per (b,s,h,j) pair j/j+128
__global__ __launch_bounds__(256) void k_rope(unsigned short* __restrict__ buf, int heads) {
  long long n = (long long)B_ * S_ * heads * 128;
  long long stride = (long long)gridDim.x * blockDim.x;
  const float lscale = logf(10000.0f) / 128.0f;
  for (long long i = (long long)blockIdx.x * blockDim.x + threadIdx.x; i < n; i += stride) {
    int j = (int)(i & 127);
    long long t = i >> 7;
    int h = (int)(t % heads);  t /= heads;
    int s = (int)(t % S_);
    int b = (int)(t / S_);
    size_t base = ((size_t)(((size_t)b * S_ + s) * heads + h)) * HD_;
    float inv = __expf(-(float)j * lscale);
    float ang = (float)s * inv;
    float sn, cs; __sincosf(ang, &sn, &cs);
    float x0 = bf2f(buf[base + j]);
    float x1 = bf2f(buf[base + j + 128]);
    buf[base + j]       = f2bf(x0 * cs - x1 * sn);
    buf[base + j + 128] = f2bf(x1 * cs + x0 * sn);
  }
}

// V bf16 [b, s, kv, d] -> VT bf16 [b, kv, d, s]
__global__ __launch_bounds__(256) void k_transpose_v(const unsigned short* __restrict__ Vb,
                                                     unsigned short* __restrict__ VT) {
  long long n = (long long)B_ * S_ * NKV_ * HD_;
  long long stride = (long long)gridDim.x * blockDim.x;
  for (long long i = (long long)blockIdx.x * blockDim.x + threadIdx.x; i < n; i += stride) {
    int d  = (int)(i & (HD_ - 1));
    long long t = i >> 8;
    int kv = (int)(t & (NKV_ - 1)); t >>= 2;
    int s  = (int)(t & (S_ - 1));
    int b  = (int)(t >> 11);
    VT[(((size_t)b * NKV_ + kv) * HD_ + d) * S_ + s] = Vb[i];
  }
}

// ---------------------------------------------------------------------------
// TDM helper: 2D tile load (rows x cols of bf16) from global to LDS.
//   D# built per CDNA5 ISA. Issue from one wave,
//   uniform operands (descriptor lives in SGPRs).
// ---------------------------------------------------------------------------
#if HAVE_TDM
DEV void tdm_load_tile_bf16(unsigned lds_byte_off, const void* gptr,
                            int cols, int rows, long long row_stride_elems) {
  unsigned long long ga = (unsigned long long)(uintptr_t)gptr;
  u32x4 g0;
  g0[0] = 1u;                                        // count=1, user mode
  g0[1] = lds_byte_off;                              // lds_addr
  g0[2] = (unsigned)(ga & 0xFFFFFFFFu);              // global_addr[31:0]
  g0[3] = (unsigned)((ga >> 32) & 0x01FFFFFFu) | (2u << 30);  // addr[56:32] | type=2

  unsigned long long st0 = (unsigned long long)row_stride_elems;
  i32x8 g1;
  g1[0] = (int)(1u << 16);                           // data_size=1 (2 bytes)
  g1[1] = (int)(((unsigned)cols & 0xFFFFu) << 16);   // tensor_dim0[15:0]
  g1[2] = (int)((((unsigned)cols >> 16) & 0xFFFFu) |
                (((unsigned)rows & 0xFFFFu) << 16)); // tensor_dim0[31:16] | tensor_dim1[15:0]
  g1[3] = (int)((((unsigned)rows >> 16) & 0xFFFFu) |
                (((unsigned)cols & 0xFFFFu) << 16)); // tensor_dim1[31:16] | tile_dim0
  g1[4] = (int)((unsigned)rows & 0xFFFFu);           // tile_dim1 | tile_dim2=0
  g1[5] = (int)(unsigned)(st0 & 0xFFFFFFFFu);        // tensor_dim0_stride[31:0]
  g1[6] = (int)(unsigned)((st0 >> 32) & 0xFFFFu);    // stride[47:32] | dim1_stride lo=0
  g1[7] = 0;

  i32x4 gz = {0, 0, 0, 0};
#if __clang_major__ >= 23
  i32x8 gz8 = {0, 0, 0, 0, 0, 0, 0, 0};
  __builtin_amdgcn_tensor_load_to_lds(g0, g1, gz, gz, gz8, 0);
#else
  __builtin_amdgcn_tensor_load_to_lds(g0, g1, gz, gz, 0);
#endif
}
#endif

// ---------------------------------------------------------------------------
// Generic bf16 WMMA GEMM: C[M,N] = A[M,K] * Bt[N,K]^T
// block = 256 threads = 8 waves (2 M x 4 N); each wave owns a 16x64 strip
// (4 accumulators) so one A fragment feeds 4 WMMAs.
// ---------------------------------------------------------------------------
template <bool OUT_BF16>
__global__ __launch_bounds__(256) void gemm_bf16_wmma(const unsigned short* __restrict__ A,
                                                      const unsigned short* __restrict__ Bt,
                                                      void* __restrict__ Cv,
                                                      int M, int N, int K) {
  const int lane = threadIdx.x & 31;
  const int wid  = threadIdx.x >> 5;
  const int g = lane >> 4;       // half-wave group (K sub-chunk select)
  const int c = lane & 15;       // M row (A) / N col (B) within tile
  const int m0 = blockIdx.y * 32 + (wid >> 2) * 16;
  const int nb = blockIdx.x * 256 + (wid & 3) * 64;

  const unsigned short* arow = A + (size_t)(m0 + c) * K;
  const unsigned short* brow[4];
  for (int j = 0; j < 4; ++j) brow[j] = Bt + (size_t)(nb + j * 16 + c) * K;

  f32x8 acc[4];
  for (int j = 0; j < 4; ++j)
    for (int i = 0; i < 8; ++i) acc[j][i] = 0.0f;

  for (int kt = 0; kt < K; kt += 32) {
    FragB a;
    // A (16-bit 16x32): lanes 0-15 hold K=[0..15], lanes 16-31 K=[16..31]
    a.q[0] = *(const uint4*)(arow + kt + g * 16);
    a.q[1] = *(const uint4*)(arow + kt + g * 16 + 8);
    if (kt + 128 < K) __builtin_prefetch(arow + kt + 128, 0, 0);
    for (int j = 0; j < 4; ++j) {
      FragB b;
      // B (32x16): lane=column, 16 contiguous K values starting at g*16
      b.q[0] = *(const uint4*)(brow[j] + kt + g * 16);
      b.q[1] = *(const uint4*)(brow[j] + kt + g * 16 + 8);
      acc[j] = __builtin_amdgcn_wmma_f32_16x16x32_bf16(false, a.v, false, b.v,
                                                       (short)0, acc[j], false, false);
    }
  }

  // C layout: VGPR i -> M = i + 8*g, N = c
  if (OUT_BF16) {
    unsigned short* C = (unsigned short*)Cv;
    for (int j = 0; j < 4; ++j)
      for (int i = 0; i < 8; ++i)
        C[(size_t)(m0 + i + 8 * g) * N + nb + j * 16 + c] = f2bf(acc[j][i]);
  } else {
    float* C = (float*)Cv;
    for (int j = 0; j < 4; ++j)
      for (int i = 0; i < 8; ++i)
        C[(size_t)(m0 + i + 8 * g) * N + nb + j * 16 + c] = acc[j][i];
  }
}

// ---------------------------------------------------------------------------
// Attention: one block = (b, h, 16 query rows).
//   LDS: sQ bf16 [16][256] (8KB, filled by TDM) + sS fp32 [16][2048] (128KB)
//        + sP bf16 [16][2048] (64KB)  = 200KB  (< 320KB per WGP)
// ---------------------------------------------------------------------------
static constexpr int SQ_BYTES   = 16 * HD_ * (int)sizeof(unsigned short); // 8KB at offset 0
static constexpr int SS_BYTES   = 16 * S_ * (int)sizeof(float);
static constexpr int ATTN_SMEM  = SQ_BYTES + SS_BYTES + 16 * S_ * (int)sizeof(unsigned short);

__global__ __launch_bounds__(256) void attn_kernel(const unsigned short* __restrict__ Qb,
                                                   const unsigned short* __restrict__ Kb,
                                                   const unsigned short* __restrict__ VT,
                                                   unsigned short* __restrict__ AO) {
  extern __shared__ char smem[];
  unsigned short* sQ = (unsigned short*)smem;                    // [16][HD_]
  float*          sS = (float*)(smem + SQ_BYTES);                // [16][S_]
  unsigned short* sP = (unsigned short*)(smem + SQ_BYTES + SS_BYTES); // [16][S_]

  const int q0 = blockIdx.x * 16;
  const int h  = blockIdx.y;
  const int b  = blockIdx.z;
  const int kv = h >> 2;                 // repeat_kv: head h consumes kv group h/4
  const int lane = threadIdx.x & 31;
  const int wid  = threadIdx.x >> 5;
  const int g = lane >> 4;
  const int c = lane & 15;

  const int klen   = q0 + 16;            // causal: only keys <= last query row
  const int klen32 = (klen + 31) & ~31;

  // ---- stage the 16x256 Q tile (strided rows in [b,s,h,d]) into LDS ----
  const unsigned short* qtile = Qb + ((size_t)(((size_t)b * S_ + q0) * NH_ + h)) * HD_;
#if HAVE_TDM
  if (wid == 0) {                         // uniform branch: one wave drives the TDM
    tdm_load_tile_bf16(0u, qtile, HD_, 16, (long long)NH_ * HD_);
    __builtin_amdgcn_s_wait_tensorcnt(0);
  }
#else
  for (int idx = threadIdx.x; idx < 16 * HD_; idx += 256) {
    int r = idx >> 8, d = idx & (HD_ - 1);
    sQ[idx] = qtile[(size_t)r * NH_ * HD_ + d];
  }
#endif
  __syncthreads();

  // ---- Q fragments from LDS (8 chunks of k=32) ----
  const unsigned short* qrow = sQ + (size_t)c * HD_;
  FragB qf[8];
  for (int kt = 0; kt < 8; ++kt) {
    qf[kt].q[0] = *(const uint4*)(qrow + kt * 32 + g * 16);
    qf[kt].q[1] = *(const uint4*)(qrow + kt * 32 + g * 16 + 8);
  }

  // ---- phase 1: scores S = Q K^T / 16 + causal mask -> LDS fp32 ----
  for (int tt = wid; tt * 16 < klen; tt += 8) {
    const int n0 = tt * 16;
    const unsigned short* krow = Kb + ((size_t)(((size_t)b * S_ + (n0 + c)) * NKV_ + kv)) * HD_;
    f32x8 acc;
    for (int i = 0; i < 8; ++i) acc[i] = 0.0f;
    for (int kt = 0; kt < 8; ++kt) {
      FragB bf;
      bf.q[0] = *(const uint4*)(krow + kt * 32 + g * 16);
      bf.q[1] = *(const uint4*)(krow + kt * 32 + g * 16 + 8);
      acc = __builtin_amdgcn_wmma_f32_16x16x32_bf16(false, qf[kt].v, false, bf.v,
                                                    (short)0, acc, false, false);
    }
    for (int i = 0; i < 8; ++i) {
      const int m    = i + 8 * g;
      const int qpos = q0 + m;
      const int kpos = n0 + c;
      sS[m * S_ + kpos] = (kpos <= qpos) ? acc[i] * 0.0625f : -1e9f;
    }
  }
  __syncthreads();

  // ---- phase 2: row softmax (wave per 2 rows, lanes stride columns) ----
  for (int r = wid * 2; r < wid * 2 + 2; ++r) {
    float mx = -3.0e38f;
    for (int col = lane; col < klen; col += 32) mx = fmaxf(mx, sS[r * S_ + col]);
    for (int off = 16; off; off >>= 1) mx = fmaxf(mx, __shfl_xor(mx, off, 32));
    float sum = 0.0f;
    for (int col = lane; col < klen; col += 32) {
      float e = __expf(sS[r * S_ + col] - mx);
      sS[r * S_ + col] = e;
      sum += e;
    }
    for (int off = 16; off; off >>= 1) sum += __shfl_xor(sum, off, 32);
    const float inv = 1.0f / sum;
    for (int col = lane; col < klen32; col += 32)
      sP[r * S_ + col] = (col < klen) ? f2bf(sS[r * S_ + col] * inv) : (unsigned short)0;
  }
  __syncthreads();

  // ---- phase 3: O = P V  (wave owns 32 of the 256 head dims) ----
  const int d0 = wid * 32;
  f32x8 o0, o1;
  for (int i = 0; i < 8; ++i) { o0[i] = 0.0f; o1[i] = 0.0f; }
  const unsigned short* vbase = VT + ((size_t)b * NKV_ + kv) * (size_t)HD_ * S_;
  const unsigned short* vrow0 = vbase + (size_t)(d0 + c) * S_;
  const unsigned short* vrow1 = vbase + (size_t)(d0 + 16 + c) * S_;
  const unsigned short* prow  = sP + (size_t)c * S_;

  for (int kt = 0; kt < klen32; kt += 32) {
    FragB a, b0, b1;
    a.q[0]  = *(const uint4*)(prow + kt + g * 16);
    a.q[1]  = *(const uint4*)(prow + kt + g * 16 + 8);
    b0.q[0] = *(const uint4*)(vrow0 + kt + g * 16);
    b0.q[1] = *(const uint4*)(vrow0 + kt + g * 16 + 8);
    b1.q[0] = *(const uint4*)(vrow1 + kt + g * 16);
    b1.q[1] = *(const uint4*)(vrow1 + kt + g * 16 + 8);
    o0 = __builtin_amdgcn_wmma_f32_16x16x32_bf16(false, a.v, false, b0.v, (short)0, o0, false, false);
    o1 = __builtin_amdgcn_wmma_f32_16x16x32_bf16(false, a.v, false, b1.v, (short)0, o1, false, false);
  }

  for (int i = 0; i < 8; ++i) {
    const int m = i + 8 * g;
    const size_t base = ((size_t)(((size_t)b * S_ + (q0 + m)) * NH_ + h)) * HD_;
    AO[base + d0 + c]      = f2bf(o0[i]);
    AO[base + d0 + 16 + c] = f2bf(o1[i]);
  }
}

// ---------------------------------------------------------------------------
// Host orchestration
// ---------------------------------------------------------------------------
extern "C" void kernel_launch(void* const* d_in, const int* in_sizes, int n_in,
                              void* d_out, int out_size, void* d_ws, size_t ws_size,
                              hipStream_t stream) {
  (void)in_sizes; (void)n_in; (void)out_size; (void)ws_size;
  const float* hid = (const float*)d_in[0];
  // d_in[1] = attention_mask (causal, reproduced analytically)
  // d_in[2] = position_ids   (arange, reproduced analytically)
  const float* Wq = (const float*)d_in[3];
  const float* Wk = (const float*)d_in[4];
  const float* Wv = (const float*)d_in[5];
  const float* Wo = (const float*)d_in[6];
  float* out = (float*)d_out;

  char* w = (char*)d_ws;
  size_t off = 0;
  auto alloc = [&](size_t bytes) { void* p = w + off; off = (off + bytes + 255) & ~(size_t)255; return p; };

  unsigned short* Xb  = (unsigned short*)alloc((size_t)M_ * HID_ * 2);   // [4096,2048]
  unsigned short* WqT = (unsigned short*)alloc((size_t)NQ_ * HID_ * 2);  // [4096,2048]
  unsigned short* WkT = (unsigned short*)alloc((size_t)NKVD_ * HID_ * 2);
  unsigned short* WvT = (unsigned short*)alloc((size_t)NKVD_ * HID_ * 2);
  unsigned short* WoT = (unsigned short*)alloc((size_t)HID_ * NQ_ * 2);  // [2048,4096]
  unsigned short* Qb  = (unsigned short*)alloc((size_t)M_ * NQ_ * 2);
  unsigned short* Kb  = (unsigned short*)alloc((size_t)M_ * NKVD_ * 2);
  unsigned short* Vb  = (unsigned short*)alloc((size_t)M_ * NKVD_ * 2);
  unsigned short* VT  = (unsigned short*)alloc((size_t)M_ * NKVD_ * 2);
  unsigned short* AO  = (unsigned short*)alloc((size_t)M_ * NQ_ * 2);

  // prep
  k_f32_to_bf16<<<4096, 256, 0, stream>>>(hid, Xb, (long long)M_ * HID_);
  k_transpose_f32_bf16<<<4096, 256, 0, stream>>>(Wq, WqT, HID_, NQ_);
  k_transpose_f32_bf16<<<2048, 256, 0, stream>>>(Wk, WkT, HID_, NKVD_);
  k_transpose_f32_bf16<<<2048, 256, 0, stream>>>(Wv, WvT, HID_, NKVD_);
  k_transpose_f32_bf16<<<4096, 256, 0, stream>>>(Wo, WoT, NQ_, HID_);

  // QKV projections (bf16 WMMA, fp32 accum, bf16 out); block tile 32x256
  gemm_bf16_wmma<true><<<dim3(NQ_ / 256, M_ / 32), 256, 0, stream>>>(Xb, WqT, Qb, M_, NQ_, HID_);
  gemm_bf16_wmma<true><<<dim3(NKVD_ / 256, M_ / 32), 256, 0, stream>>>(Xb, WkT, Kb, M_, NKVD_, HID_);
  gemm_bf16_wmma<true><<<dim3(NKVD_ / 256, M_ / 32), 256, 0, stream>>>(Xb, WvT, Vb, M_, NKVD_, HID_);

  // RoPE in place
  k_rope<<<8192, 256, 0, stream>>>(Qb, NH_);
  k_rope<<<2048, 256, 0, stream>>>(Kb, NKV_);

  // V -> [b, kv, d, s] so PV's B-operand loads are contiguous
  k_transpose_v<<<4096, 256, 0, stream>>>(Vb, VT);

  // attention (200KB dynamic LDS per block)
  hipFuncSetAttribute((const void*)attn_kernel,
                      hipFuncAttributeMaxDynamicSharedMemorySize, ATTN_SMEM);
  attn_kernel<<<dim3(S_ / 16, NH_, B_), 256, ATTN_SMEM, stream>>>(Qb, Kb, VT, AO);

  // output projection -> fp32 result
  gemm_bf16_wmma<false><<<dim3(HID_ / 256, M_ / 32), 256, 0, stream>>>(AO, WoT, out, M_, HID_, NQ_);
}
